// AttrModel_55448027791636
// MI455X (gfx1250) — matrix operand order repaired and checked by reading, not compiled, not run on hardware
//
#include <hip/hip_runtime.h>

typedef __attribute__((ext_vector_type(16))) _Float16 v16h;
typedef __attribute__((ext_vector_type(8)))  float    v8f;

#define WAVES_PER_BLOCK 8
#define TILE_M 16

__global__ void zero_out_kernel(float* out) { out[0] = 0.0f; }

__device__ __forceinline__ int lower_bound_i32(const int* __restrict__ arr, int n, int key) {
  int lo = 0, hi = n;
  while (lo < hi) {
    int mid = (lo + hi) >> 1;
    if (arr[mid] < key) lo = mid + 1; else hi = mid;
  }
  return lo;
}

__launch_bounds__(256)
__global__ void attr_loss_kernel(
    const int*   __restrict__ char_ids,
    const int*   __restrict__ segment_ids,
    const int*   __restrict__ head_ids,
    const int*   __restrict__ rel_ids,
    const float* __restrict__ char_emb,   // [128][64]
    const float* __restrict__ rel_emb,    // [n_rel][64]
    const float* __restrict__ ent_emb,    // [n_ent][64]
    float*       __restrict__ out,
    int n_triples, int total_chars, int n_rel)
{
  // char table transposed to K-major f16 with +8 half padding per row (bank-conflict free)
  __shared__ _Float16 s_embT[64 * 136];                    // 17408 B
  __shared__ float    s_rel[22 * 64];                      //  5632 B
  __shared__ unsigned s_cnt[WAVES_PER_BLOCK * 16 * 64];    // 32768 B (packed u16 pairs)

  const int tid = threadIdx.x;

  // cooperative table staging (fp32 -> f16 convert for the WMMA B operand)
  for (int i = tid; i < 128 * 64; i += blockDim.x) {
    int k = i >> 6, n = i & 63;
    s_embT[n * 136 + k] = (_Float16)char_emb[i];
  }
  for (int i = tid; i < n_rel * 64 && i < 22 * 64; i += blockDim.x)
    s_rel[i] = rel_emb[i];
  __syncthreads();

  const int wave = tid >> 5;
  const int lane = tid & 31;
  const int numTiles = (n_triples + TILE_M - 1) / TILE_M;
  const int tile = blockIdx.x * WAVES_PER_BLOCK + wave;
  if (tile >= numTiles) return;                 // uniform per wave: EXEC stays all-1s for WMMA
  const int t0 = tile * TILE_M;

  // ---- zero this wave's histogram region
  unsigned* cnt = s_cnt + wave * (16 * 64);
  for (int i = lane; i < 16 * 64; i += 32) cnt[i] = 0u;
  asm volatile("s_wait_dscnt 0x0" ::: "memory");

  // ---- char range for this tile (lanes 0-15 find start, 16-31 find end; sorted segment_ids)
  int key = t0 + ((lane >> 4) << 4);
  int lb  = lower_bound_i32(segment_ids, total_chars, key);
  int tileStart = __shfl(lb, 0, 32);
  int tileEnd   = __shfl(lb, 16, 32);

  // ---- histogram: counts[row][char] as packed u16 pairs in u32 (LDS atomics; no carry risk)
  for (int i = tileStart + lane; i < tileEnd; i += 32) {
    int c = char_ids[i];
    int m = segment_ids[i] - t0;                // 0..15
    atomicAdd(&cnt[m * 64 + (c >> 1)], 1u << ((c & 1) << 4));
  }
  asm volatile("s_wait_dscnt 0x0" ::: "memory");

  // ---- preload ids for the rows this lane-half owns in the C layout
  const int mbase = (lane >> 4) << 3;           // 0 or 8
  int hids[8], rids[8];
#pragma unroll
  for (int r = 0; r < 8; ++r) {
    int trip = t0 + mbase + r;
    if (trip >= n_triples) trip = n_triples - 1;   // clamp (contribution masked later)
    hids[r] = head_ids[trip];
    rids[r] = rel_ids[trip];
  }

  const unsigned* crow = cnt + (lane & 15) * 64;   // A row (M = lane&15)
  const int kh2   = (lane < 16) ? 0 : 4;           // A: khalf/2 per ISA 16-bit A 16x32 layout
  const int koffB = (lane < 16) ? 0 : 8;           // B: K offset per lane half

  // ---- 16x128 @ 128x64 GEMM: kb-outer so each A fragment (u16->f16 convert) is built ONCE
  v8f acc[4];
#pragma unroll
  for (int nt = 0; nt < 4; ++nt) acc[nt] = (v8f){0.f,0.f,0.f,0.f,0.f,0.f,0.f,0.f};

#pragma unroll
  for (int kb = 0; kb < 4; ++kb) {               // K = 128 in 4 steps of 32
    // A: VGPR j holds K = kb*32 + khalf + (j<4 ? 2j : 16+2(j-4)) .. +1
    v16h A;
#pragma unroll
    for (int j = 0; j < 8; ++j) {
      int word = kb * 16 + kh2 + ((j < 4) ? j : (4 + j));
      unsigned w = crow[word];
      A[2 * j]     = (_Float16)(float)(w & 0xFFFFu);
      A[2 * j + 1] = (_Float16)(float)(w >> 16);
    }
#pragma unroll
    for (int nt = 0; nt < 4; ++nt) {             // 16-wide N slabs covering D=64
      const _Float16* bp = s_embT + (nt * 16 + (lane & 15)) * 136;
      v16h B;
#pragma unroll
      for (int j = 0; j < 8; ++j) {              // B: VGPR j holds K = kb*32 + koffB + (j<4 ? 2j : 16+2(j-4)), +1
        int K = kb * 32 + koffB + ((j < 4) ? 2 * j : 8 + 2 * j);
        B[2 * j]     = bp[K];
        B[2 * j + 1] = bp[K + 1];
      }
      acc[nt] = __builtin_amdgcn_wmma_f32_16x16x32_f16(
          /*neg_a=*/false, A, /*neg_b=*/false, B,
          /*c_mod=*/(short)0, acc[nt], /*reuse_a=*/false, /*reuse_b=*/false);
    }
  }

  // ---- fused epilogue on accumulator registers: |h + r - t|
  float rowsum[8];
#pragma unroll
  for (int r = 0; r < 8; ++r) rowsum[r] = 0.f;

#pragma unroll
  for (int nt = 0; nt < 4; ++nt) {
    int nn = nt * 16 + (lane & 15);
#pragma unroll
    for (int r = 0; r < 8; ++r) {
      float h  = ent_emb[hids[r] * 64 + nn];
      float rv = s_rel[rids[r] * 64 + nn];
      rowsum[r] += fabsf(h + rv - acc[nt][r]);
    }
  }

  // ---- reduce each row across its 16-lane half, apply ReLU(dist + GAMMA), accumulate
  float waveLoss = 0.f;
#pragma unroll
  for (int r = 0; r < 8; ++r) {
    float v = rowsum[r];
#pragma unroll
    for (int off = 1; off < 16; off <<= 1) v += __shfl_xor(v, off, 32);
    if (((lane & 15) == 0) && (t0 + mbase + r) < n_triples)
      waveLoss += fmaxf(v + 1.0f, 0.f);            // GAMMA = 1.0
  }
  waveLoss += __shfl_xor(waveLoss, 16, 32);        // combine lane 0 (rows 0-7) + lane 16 (rows 8-15)
  if (lane == 0) atomicAdd(out, waveLoss);
}

extern "C" void kernel_launch(void* const* d_in, const int* in_sizes, int n_in,
                              void* d_out, int out_size, void* d_ws, size_t ws_size,
                              hipStream_t stream) {
  const int*   char_ids    = (const int*)d_in[0];
  const int*   segment_ids = (const int*)d_in[1];
  const int*   head_ids    = (const int*)d_in[2];
  const int*   rel_ids     = (const int*)d_in[3];
  const float* char_emb    = (const float*)d_in[4];
  const float* rel_emb     = (const float*)d_in[5];
  const float* ent_emb     = (const float*)d_in[6];
  float* out = (float*)d_out;

  const int total_chars = in_sizes[0];
  const int n_triples   = in_sizes[2];
  const int n_rel       = in_sizes[5] / 64;

  const int numTiles = (n_triples + TILE_M - 1) / TILE_M;
  const int blocks   = (numTiles + WAVES_PER_BLOCK - 1) / WAVES_PER_BLOCK;

  hipLaunchKernelGGL(zero_out_kernel, dim3(1), dim3(1), 0, stream, out);
  hipLaunchKernelGGL(attr_loss_kernel, dim3(blocks), dim3(256), 0, stream,
                     char_ids, segment_ids, head_ids, rel_ids,
                     char_emb, rel_emb, ent_emb, out,
                     n_triples, total_chars, n_rel);
}
